// CluTSPEncoder_7069516169274
// MI455X (gfx1250) — matrix operand order repaired and checked by reading, not compiled, not run on hardware
//
#include <hip/hip_runtime.h>
#include <hip/hip_bf16.h>

// ---------------------------------------------------------------------------
// CluTSP encoder for gfx1250 (MI455X): bf16 WMMA GEMMs + flash-style GATv2.
// B=32, N=256, D=128, H=8, Q=16, FF=512, L=3.
// GEMM: block tile 128x64, 8 waves, 4 WMMA accumulators/wave, LDS double-buffer.
// ---------------------------------------------------------------------------

typedef __bf16 bf16;
typedef bf16  bf16x16 __attribute__((ext_vector_type(16)));
typedef bf16  bf16x8  __attribute__((ext_vector_type(8)));
typedef float floatx8 __attribute__((ext_vector_type(8)));

union AF { bf16x16 v; bf16x8 h[2]; };

#define NEG_SLOPE 0.15f
#define MASK_VAL  -1.0e9f
#define BNROWS    8192       // B*N
#define DM        128
#define HH        8
#define QQ        16
#define FFM       512
#define LM        3

static __device__ __forceinline__ float lrelu(float x) {
    return x > 0.0f ? x : NEG_SLOPE * x;
}

// ---------------------------------------------------------------------------
// fp32 -> bf16 conversion (weights, once per launch)
// ---------------------------------------------------------------------------
__global__ void k_cvt(const float* __restrict__ x, bf16* __restrict__ y, int n) {
    int t = blockIdx.x * blockDim.x + threadIdx.x;
    if (t < n) y[t] = (bf16)x[t];
}

// ---------------------------------------------------------------------------
// h0 = pos[:, :16] @ W_init.T + b_init ; also emit bf16 copy and cluster ids
// ---------------------------------------------------------------------------
__global__ void k_init(const float* __restrict__ pos, const float* __restrict__ W,
                       const float* __restrict__ b, float* __restrict__ h,
                       bf16* __restrict__ hbf, int* __restrict__ cluster) {
    int t = blockIdx.x * blockDim.x + threadIdx.x;
    if (t >= BNROWS * DM) return;
    int d = t & (DM - 1), bn = t >> 7;
    const float* pr = pos + (size_t)bn * 17;
    const float* wr = W + (size_t)d * 16;
    float acc = b[d];
#pragma unroll
    for (int k = 0; k < 16; ++k) acc += pr[k] * wr[k];
    h[t]   = acc;
    hbf[t] = (bf16)acc;
    if (d == 0) cluster[bn] = (int)pr[16];
}

// ---------------------------------------------------------------------------
// si/sj head projections: si[bn,h] = h_row . Wa[h, 0:128], sj = . Wa[h,128:256]
// ---------------------------------------------------------------------------
__global__ void k_sisj(const float* __restrict__ h, const float* __restrict__ Wa,
                       int wstride, float* __restrict__ si, float* __restrict__ sj) {
    int t = blockIdx.x * blockDim.x + threadIdx.x;
    if (t >= BNROWS * HH) return;
    int head = t & (HH - 1), bn = t >> 3;
    const float* hr = h + (size_t)bn * DM;
    const float* wi = Wa + (size_t)head * wstride;
    const float* wj = wi + DM;
    float a = 0.0f, bsum = 0.0f;
#pragma unroll 4
    for (int k = 0; k < DM; ++k) { float hv = hr[k]; a += hv * wi[k]; bsum += hv * wj[k]; }
    si[(size_t)bn * HH + head] = a;
    sj[(size_t)bn * HH + head] = bsum;
}

// ---------------------------------------------------------------------------
// pack v fp32 [bn, h*16+q] -> vb bf16 [b][h][q][n]  (B-matrix layout for attn)
// ---------------------------------------------------------------------------
__global__ void k_pack_v(const float* __restrict__ v, bf16* __restrict__ vb) {
    int t = blockIdx.x * blockDim.x + threadIdx.x;
    if (t >= BNROWS * DM) return;
    int hq = t & (DM - 1), bn = t >> 7;
    int b = bn >> 8, n = bn & 255;
    int hd = hq >> 4, q = hq & 15;
    vb[(((size_t)(b * HH + hd) * QQ + q) * 256) + n] = (bf16)v[t];
}

// ---------------------------------------------------------------------------
// Generic bf16 WMMA GEMM: C[M,N] = A[M,K] @ W[N,K]^T (+bias)(+ELU)
// block = 256 thr (8 waves). Block tile 128(M) x 64(N), K stepped by 32.
// Each wave owns a 16x64 C strip = 4 WMMA accumulators; A fragment loaded once
// per K-step and reused 4x. LDS double-buffered: one barrier per K-step, the
// global->LDS loads for step k+32 overlap the 4 WMMAs of step k.
// Requires: M % 128 == 0, N % 64 == 0, K % 32 == 0.
// ---------------------------------------------------------------------------
__launch_bounds__(256)
__global__ void k_gemm(const bf16* __restrict__ A, const bf16* __restrict__ W,
                       float* __restrict__ C, bf16* __restrict__ Cbf,
                       const float* __restrict__ bias,
                       int M, int N, int K, int act, int cbf_ld, int cbf_off) {
    __shared__ __align__(16) bf16 ldsA[2][8][16 * 32];   // per-wave 16x32 tiles
    __shared__ __align__(16) bf16 ldsB[2][64 * 32];      // 64(n) x 32(k) weights
    int tid = threadIdx.x;
    int wv = tid >> 5, lane = tid & 31;
    int n0 = blockIdx.x * 64, m0 = blockIdx.y * 128;
    int half = lane >> 4, rc = lane & 15;

    floatx8 acc[4];
#pragma unroll
    for (int s = 0; s < 4; ++s)
        acc[s] = (floatx8){0.f, 0.f, 0.f, 0.f, 0.f, 0.f, 0.f, 0.f};

    auto load_tiles = [&](int k0, int buf) {
        // B: 64 rows x 32 k = 256 x 16B chunks, one per thread
        {
            int r = tid >> 2, c = tid & 3;
            *(bf16x8*)&ldsB[buf][r * 32 + c * 8] =
                *(const bf16x8*)&W[(size_t)(n0 + r) * K + k0 + c * 8];
        }
        // A: per-wave 16x32 tile; 2 x 16B chunks per lane
        {
            const bf16* Arow = &A[(size_t)(m0 + wv * 16) * K + k0];
            int c0 = lane * 2;
            int r0 = c0 >> 2, cc0 = c0 & 3;
            *(bf16x8*)&ldsA[buf][wv][r0 * 32 + cc0 * 8] =
                *(const bf16x8*)&Arow[(size_t)r0 * K + cc0 * 8];
            int c1 = c0 + 1;
            int r1 = c1 >> 2, cc1 = c1 & 3;
            *(bf16x8*)&ldsA[buf][wv][r1 * 32 + cc1 * 8] =
                *(const bf16x8*)&Arow[(size_t)r1 * K + cc1 * 8];
        }
    };

    load_tiles(0, 0);
    __syncthreads();

    for (int k0 = 0; k0 < K; k0 += 32) {
        int buf = (k0 >> 5) & 1;
        if (k0 + 32 < K) load_tiles(k0 + 32, buf ^ 1);  // overlap with WMMAs

        AF a;  // A fragment, reused for all 4 n-subtiles
        a.h[0] = *(const bf16x8*)&ldsA[buf][wv][rc * 32 + half * 8];
        a.h[1] = *(const bf16x8*)&ldsA[buf][wv][rc * 32 + 16 + half * 8];
#pragma unroll
        for (int s = 0; s < 4; ++s) {
            AF bfr;
            bfr.h[0] = *(const bf16x8*)&ldsB[buf][(s * 16 + rc) * 32 + half * 8];
            bfr.h[1] = *(const bf16x8*)&ldsB[buf][(s * 16 + rc) * 32 + 16 + half * 8];
            acc[s] = __builtin_amdgcn_wmma_f32_16x16x32_bf16(
                false, a.v, false, bfr.v, (short)0, acc[s], false, false);
        }
        __syncthreads();
    }

    // --- epilogue: bias, optional ELU, fp32 + optional bf16 store
#pragma unroll
    for (int s = 0; s < 4; ++s) {
        int n = n0 + s * 16 + rc;
        float bv = bias ? bias[n] : 0.0f;
#pragma unroll
        for (int r = 0; r < 8; ++r) {
            int m = m0 + wv * 16 + r + 8 * half;
            float val = acc[s][r] + bv;
            if (act == 1) val = val > 0.0f ? val : __expf(val) - 1.0f;  // ELU
            if (C)   C[(size_t)m * N + n] = val;
            if (Cbf) Cbf[(size_t)m * cbf_ld + cbf_off + n] = (bf16)val;
        }
    }
}

// ---------------------------------------------------------------------------
// Flash-style GATv2 attention. One wave per (b, head, 16-row i-tile).
// scores(i,j) = lrelu(si[i]+sj[j]+same*we), optional local mask; softmax over j
// via O(N)-per-row stats (rank-1 score structure), alpha tiles built in
// registers (bf16) and contracted against V with 8x WMMA 16x16x32.
// ---------------------------------------------------------------------------
__launch_bounds__(32)
__global__ void k_attn(const float* __restrict__ si, const float* __restrict__ sj,
                       const bf16* __restrict__ vb, const int* __restrict__ cluster,
                       const float* __restrict__ Wa, int wa_stride,
                       int use_edge, int local_only, bf16* __restrict__ attn_bf) {
    int lane  = threadIdx.x;
    int itile = blockIdx.x, hd = blockIdx.y, b = blockIdx.z;
    int i0 = itile * 16;

    __shared__ float sjr[256];
    __shared__ int   cl[256];
    __shared__ float sir[16], mrow[16], rdrow[16];

    for (int j = lane; j < 256; j += 32) {
        sjr[j] = sj[(size_t)(b * 256 + j) * HH + hd];
        cl[j]  = cluster[b * 256 + j];
    }
    if (lane < 16) sir[lane] = si[(size_t)(b * 256 + i0 + lane) * HH + hd];
    float we = use_edge ? Wa[(size_t)hd * wa_stride + 2 * DM] : 0.0f;
    __syncthreads();

    // pass 1: per-row softmax stats (lanes 0..15, one row each)
    if (lane < 16) {
        int ci = cl[i0 + lane];
        float siv = sir[lane];
        float m = -3.0e38f;
        for (int j = 0; j < 256; ++j) {
            float s = siv + sjr[j];
            if (use_edge && cl[j] == ci) s += we;
            s = lrelu(s);
            if (local_only && cl[j] != ci) s = MASK_VAL;
            m = fmaxf(m, s);
        }
        float sum = 0.0f;
        for (int j = 0; j < 256; ++j) {
            float s = siv + sjr[j];
            if (use_edge && cl[j] == ci) s += we;
            s = lrelu(s);
            if (local_only && cl[j] != ci) s = MASK_VAL;
            sum += __expf(s - m);
        }
        mrow[lane]  = m;
        rdrow[lane] = 1.0f / sum;
    }
    __syncthreads();

    // pass 2: alpha tiles -> WMMA against V  (EXEC all-ones here, no divergence)
    int half = lane >> 4, row = lane & 15;
    float m = mrow[row], rd = rdrow[row], siv = sir[row];
    int ci = cl[i0 + row];
    const bf16* vrow = vb + ((size_t)(b * HH + hd) * QQ + row) * 256;  // [q][j]

    floatx8 acc = {0.f, 0.f, 0.f, 0.f, 0.f, 0.f, 0.f, 0.f};
    for (int kt = 0; kt < 8; ++kt) {
        int j0 = kt * 32;
        AF a;
#pragma unroll
        for (int e = 0; e < 16; ++e) {
            int j = j0 + (e >> 3) * 16 + half * 8 + (e & 7);
            float s = siv + sjr[j];
            if (use_edge && cl[j] == ci) s += we;
            s = lrelu(s);
            if (local_only && cl[j] != ci) s = MASK_VAL;
            a.v[e] = (bf16)(__expf(s - m) * rd);
        }
        AF bfr;  // B frag: n = q = row; K in table order (half*8, 16+half*8)
        bfr.h[0] = *(const bf16x8*)(vrow + j0 + half * 8);
        bfr.h[1] = *(const bf16x8*)(vrow + j0 + 16 + half * 8);
        acc = __builtin_amdgcn_wmma_f32_16x16x32_bf16(
            false, a.v, false, bfr.v, (short)0, acc, false, false);
    }

    // store bf16 attn output at [bn][h*16+q], ready as GEMM A-operand for Wo
#pragma unroll
    for (int r = 0; r < 8; ++r) {
        int i = i0 + r + 8 * half;
        attn_bf[((size_t)(b * 256 + i)) * DM + hd * QQ + row] = (bf16)acc[r];
    }
}

// ---------------------------------------------------------------------------
// BatchNorm (training mode, over B*N rows) with residual: y = BN(x + res)
// one block per feature d; writes fp32 + bf16 (strided, for concat halves)
// ---------------------------------------------------------------------------
__launch_bounds__(256)
__global__ void k_bn(const float* __restrict__ x, const float* __restrict__ res,
                     const float* __restrict__ gamma, const float* __restrict__ beta,
                     float* __restrict__ yout, bf16* __restrict__ ybf,
                     int ybf_ld, int ybf_off) {
    int d = blockIdx.x, tid = threadIdx.x;
    __shared__ float ssum[256], ssq[256];
    float s = 0.0f, s2 = 0.0f;
    for (int r = tid; r < BNROWS; r += 256) {
        float v = x[(size_t)r * DM + d] + res[(size_t)r * DM + d];
        s += v; s2 += v * v;
    }
    ssum[tid] = s; ssq[tid] = s2;
    __syncthreads();
    for (int off = 128; off > 0; off >>= 1) {
        if (tid < off) { ssum[tid] += ssum[tid + off]; ssq[tid] += ssq[tid + off]; }
        __syncthreads();
    }
    float mean = ssum[0] * (1.0f / BNROWS);
    float var  = ssq[0] * (1.0f / BNROWS) - mean * mean;
    float rs   = rsqrtf(var + 1e-5f);
    float g = gamma[d], be = beta[d];
    for (int r = tid; r < BNROWS; r += 256) {
        float v = x[(size_t)r * DM + d] + res[(size_t)r * DM + d];
        float y = (v - mean) * rs * g + be;
        if (yout) yout[(size_t)r * DM + d] = y;
        ybf[(size_t)r * ybf_ld + ybf_off + d] = (bf16)y;
    }
}

// ---------------------------------------------------------------------------
// host driver
// ---------------------------------------------------------------------------
extern "C" void kernel_launch(void* const* d_in, const int* in_sizes, int n_in,
                              void* d_out, int out_size, void* d_ws, size_t ws_size,
                              hipStream_t stream) {
    const float* pos    = (const float*)d_in[0];
    const float* Winit  = (const float*)d_in[1];
    const float* binit  = (const float*)d_in[2];
    const float* Wa_g   = (const float*)d_in[3];
    const float* Wv_g   = (const float*)d_in[4];
    const float* Wo_g   = (const float*)d_in[5];
    const float* Wa_l   = (const float*)d_in[6];
    const float* Wv_l   = (const float*)d_in[7];
    const float* Wo_l   = (const float*)d_in[8];
    const float* bng    = (const float*)d_in[9];
    const float* bnb    = (const float*)d_in[10];
    const float* ff1w_g = (const float*)d_in[11];
    const float* ff1b_g = (const float*)d_in[12];
    const float* ff2w_g = (const float*)d_in[13];
    const float* ff2b_g = (const float*)d_in[14];
    const float* ff1w_l = (const float*)d_in[15];
    const float* ff1b_l = (const float*)d_in[16];
    const float* ff2w_l = (const float*)d_in[17];
    const float* ff2b_l = (const float*)d_in[18];
    const float* Wfuse  = (const float*)d_in[19];
    const float* bfuse  = (const float*)d_in[20];

    char* wp = (char*)d_ws;
    auto alloc = [&](size_t bytes) -> void* {
        void* p = (void*)wp;
        wp += (bytes + 255) & ~(size_t)255;
        return p;
    };

    float* h       = (float*)alloc((size_t)BNROWS * DM * 4);
    bf16*  hbf     = (bf16*)alloc((size_t)BNROWS * DM * 2);
    int*   cluster = (int*)alloc((size_t)BNROWS * 4);
    float* si      = (float*)alloc((size_t)BNROWS * HH * 4);
    float* sj      = (float*)alloc((size_t)BNROWS * HH * 4);
    float* v       = (float*)alloc((size_t)BNROWS * DM * 4);
    bf16*  vb      = (bf16*)alloc((size_t)BNROWS * DM * 2);
    bf16*  attnbf  = (bf16*)alloc((size_t)BNROWS * DM * 2);
    float* gpre    = (float*)alloc((size_t)BNROWS * DM * 4);
    float* gbn     = (float*)alloc((size_t)BNROWS * DM * 4);
    bf16*  gbnbf   = (bf16*)alloc((size_t)BNROWS * DM * 2);
    float* ffmid   = (float*)alloc((size_t)BNROWS * FFM * 4);
    bf16*  ffmidbf = (bf16*)alloc((size_t)BNROWS * FFM * 2);
    float* ff2o    = (float*)alloc((size_t)BNROWS * DM * 4);
    float* bn2f    = (float*)alloc((size_t)BNROWS * DM * 4);
    bf16*  concatb = (bf16*)alloc((size_t)BNROWS * 2 * DM * 2);

    bf16* Wv_g_bf   = (bf16*)alloc((size_t)LM * DM * DM * 2);
    bf16* Wo_g_bf   = (bf16*)alloc((size_t)LM * DM * DM * 2);
    bf16* Wv_l_bf   = (bf16*)alloc((size_t)LM * DM * DM * 2);
    bf16* Wo_l_bf   = (bf16*)alloc((size_t)LM * DM * DM * 2);
    bf16* ff1wg_bf  = (bf16*)alloc((size_t)LM * FFM * DM * 2);
    bf16* ff2wg_bf  = (bf16*)alloc((size_t)LM * DM * FFM * 2);
    bf16* ff1wl_bf  = (bf16*)alloc((size_t)LM * FFM * DM * 2);
    bf16* ff2wl_bf  = (bf16*)alloc((size_t)LM * DM * FFM * 2);
    bf16* Wfuse_bf  = (bf16*)alloc((size_t)LM * DM * 2 * DM * 2);

    auto cvt = [&](const float* src, bf16* dst, int n) {
        k_cvt<<<(n + 255) / 256, 256, 0, stream>>>(src, dst, n);
    };
    cvt(Wv_g, Wv_g_bf, LM * DM * DM);
    cvt(Wo_g, Wo_g_bf, LM * DM * DM);
    cvt(Wv_l, Wv_l_bf, LM * DM * DM);
    cvt(Wo_l, Wo_l_bf, LM * DM * DM);
    cvt(ff1w_g, ff1wg_bf, LM * FFM * DM);
    cvt(ff2w_g, ff2wg_bf, LM * DM * FFM);
    cvt(ff1w_l, ff1wl_bf, LM * FFM * DM);
    cvt(ff2w_l, ff2wl_bf, LM * DM * FFM);
    cvt(Wfuse, Wfuse_bf, LM * DM * 2 * DM);

    k_init<<<(BNROWS * DM + 255) / 256, 256, 0, stream>>>(pos, Winit, binit, h, hbf, cluster);

    for (int l = 0; l < LM; ++l) {
        for (int t = 0; t < 2; ++t) {  // t=0 global (edge feature), t=1 local (masked)
            int wstride = (t == 0) ? (2 * DM + 1) : (2 * DM);
            const float* Wa = (t == 0) ? (Wa_g + (size_t)l * HH * (2 * DM + 1))
                                       : (Wa_l + (size_t)l * HH * (2 * DM));
            const bf16* Wv  = ((t == 0) ? Wv_g_bf : Wv_l_bf) + (size_t)l * DM * DM;
            const bf16* Wo  = ((t == 0) ? Wo_g_bf : Wo_l_bf) + (size_t)l * DM * DM;
            const bf16* f1w = ((t == 0) ? ff1wg_bf : ff1wl_bf) + (size_t)l * FFM * DM;
            const bf16* f2w = ((t == 0) ? ff2wg_bf : ff2wl_bf) + (size_t)l * DM * FFM;
            const float* f1b = ((t == 0) ? ff1b_g : ff1b_l) + (size_t)l * FFM;
            const float* f2b = ((t == 0) ? ff2b_g : ff2b_l) + (size_t)l * DM;
            const float* g0 = bng + ((size_t)l * 4 + (t == 0 ? 0 : 2)) * DM;
            const float* b0 = bnb + ((size_t)l * 4 + (t == 0 ? 0 : 2)) * DM;
            const float* g1 = bng + ((size_t)l * 4 + (t == 0 ? 1 : 3)) * DM;
            const float* b1 = bnb + ((size_t)l * 4 + (t == 0 ? 1 : 3)) * DM;

            k_sisj<<<(BNROWS * HH + 255) / 256, 256, 0, stream>>>(h, Wa, wstride, si, sj);
            // v = h @ Wv.T                      (N=128 -> grid.x = 2)
            k_gemm<<<dim3(DM / 64, BNROWS / 128), 256, 0, stream>>>(
                hbf, Wv, v, nullptr, nullptr, BNROWS, DM, DM, 0, 0, 0);
            k_pack_v<<<(BNROWS * DM + 255) / 256, 256, 0, stream>>>(v, vb);
            // flash-GAT: softmax(alpha) @ v  (WMMA)
            k_attn<<<dim3(256 / 16, HH, 32), 32, 0, stream>>>(
                si, sj, vb, cluster, Wa, wstride,
                /*use_edge=*/(t == 0) ? 1 : 0, /*local_only=*/(t == 1) ? 1 : 0, attnbf);
            // out @ Wo.T
            k_gemm<<<dim3(DM / 64, BNROWS / 128), 256, 0, stream>>>(
                attnbf, Wo, gpre, nullptr, nullptr, BNROWS, DM, DM, 0, 0, 0);
            // BN(gat + h)
            k_bn<<<DM, 256, 0, stream>>>(gpre, h, g0, b0, gbn, gbnbf, DM, 0);
            // FF1 + ELU (fused epilogue)     (N=512 -> grid.x = 8)
            k_gemm<<<dim3(FFM / 64, BNROWS / 128), 256, 0, stream>>>(
                gbnbf, f1w, ffmid, ffmidbf, f1b, BNROWS, FFM, DM, 1, FFM, 0);
            // FF2
            k_gemm<<<dim3(DM / 64, BNROWS / 128), 256, 0, stream>>>(
                ffmidbf, f2w, ff2o, nullptr, f2b, BNROWS, DM, FFM, 0, 0, 0);
            // BN(gbn + ff2) -> bf16 into concat half [.,t*128:t*128+128]
            k_bn<<<DM, 256, 0, stream>>>(ff2o, gbn, g1, b1, bn2f, concatb, 2 * DM, t * DM);
        }
        // fuse: concat(g,l) @ Wfuse.T + bfuse -> h (last layer -> d_out)
        const bf16* Wf = Wfuse_bf + (size_t)l * DM * 2 * DM;
        const float* bfv = bfuse + (size_t)l * DM;
        float* Cout = (l == LM - 1) ? (float*)d_out : h;
        k_gemm<<<dim3(DM / 64, BNROWS / 128), 256, 0, stream>>>(
            concatb, Wf, Cout, hbf, bfv, BNROWS, DM, 2 * DM, 0, DM, 0);
    }
}
